// TransitionGNN_46093589021064
// MI455X (gfx1250) — matrix operand order repaired and checked by reading, not compiled, not run on hardware
//
#include <hip/hip_runtime.h>

// ---------------- problem constants ----------------
#define NB       1024
#define NOBJ     10
#define DIN      32
#define HID      512
#define ADIM     4
#define NEDGE    90           // NOBJ*(NOBJ-1)
#define EPAD     96           // edges padded to 6 M-tiles of 16
#define NIN_PAD  576          // 548 padded to 18 K-tiles of 32
#define WAVES    8

typedef __bf16 bf16_t;
typedef __attribute__((ext_vector_type(16))) __bf16 v16bf;
typedef __attribute__((ext_vector_type(8)))  __bf16 v8bf;
typedef __attribute__((ext_vector_type(8)))  float  v8f;

// ---------------- workspace layout (bf16 elements) ----------------
// Weights pre-swizzled into WMMA B-fragment blocks of 512 bf16 (32 lanes x 16).
#define OFF_EW1  0          // K=64  (Kt=2),  N=512 (Nt=32) -> 32768
#define OFF_EW2  32768      // K=512 (Kt=16), N=512 (Nt=32) -> 262144
#define OFF_EW3  294912     // same                          -> 262144
#define OFF_NW1  557056     // K=548->576 (Kt=18), N=512     -> 294912
#define OFF_NW2  851968     // K=512, N=512                  -> 262144
#define OFF_NW3  1114112    // K=512 (Kt=16), N=32 (Nt=2)    -> 16384
#define WS_ELEMS 1130496

// ---------------- LDS layout (bytes) ----------------
#define LDS_EDGEIN  0        // [96][64]  bf16 = 12288
#define LDS_ACT     12288    // [96][512] bf16 = 98304
#define LDS_ZBUF    110592   // [96][512] f32  = 196608
#define LDS_RMEAN   307200   // [96] f32
#define LDS_RRSTD   307584   // [96] f32
#define LDS_BYTES   307968
#define LDS_NODEIN  12288    // [16][576] bf16 (reuses ACT after edge GEMM3)
#define LDS_NACT    30720    // [16][512] bf16 (reuses ACT)

// ---------------- weight swizzle: f32 row-major -> bf16 B-fragments ----------------
__global__ __launch_bounds__(256) void swizzle_w_kernel(
    const float* __restrict__ W, bf16_t* __restrict__ dst,
    int Kdim, int Ndim, int Nt, int total) {
  int idx = blockIdx.x * 256 + threadIdx.x;
  if (idx >= total) return;
  int blk  = idx >> 9;           // fragment block (kt*Nt + nt)
  int r    = idx & 511;
  int lane = r >> 4;
  int e    = r & 15;
  int kt = blk / Nt, nt = blk - kt * Nt;
  int n = nt * 16 + (lane & 15);
  int k = kt * 32 + ((e >> 3) << 4) + ((lane >> 4) << 3) + (e & 7);   // lane's 16 elems = K [8g..8g+7],[16+8g..16+8g+7]
  float v = (k < Kdim && n < Ndim) ? W[(size_t)k * Ndim + n] : 0.0f;
  dst[idx] = (bf16_t)v;
}

// ---------------- fragment loads ----------------
// A fragment (16x32 bf16 tile from LDS): lane l holds row m = l&15,
// K elems [8g..8g+7] and [16+8g..16+8g+7], g = l>>4  (ISA 7.12.2)
__device__ __forceinline__ v16bf load_a_frag(const bf16_t* base, int stride,
                                             int mt, int kt, int lane) {
  int g = lane >> 4;
  const bf16_t* row = base + (mt * 16 + (lane & 15)) * stride + kt * 32;
  v8bf lo = *(const v8bf*)(row + 8 * g);
  v8bf hi = *(const v8bf*)(row + 16 + 8 * g);
  v16bf a;
#pragma unroll
  for (int i = 0; i < 8; ++i) { a[i] = lo[i]; a[i + 8] = hi[i]; }
  return a;
}

// B fragment: pre-swizzled in global; lane reads 16 contiguous bf16 (32B).
__device__ __forceinline__ v16bf load_b_frag(const bf16_t* wsw, int Nt,
                                             int kt, int nt, int lane) {
  return *(const v16bf*)(wsw + (((size_t)(kt * Nt + nt)) << 9) + (lane << 4));
}

// 16x32 output tile: one A fragment feeds two back-to-back WMMAs.
__device__ __forceinline__ void mm_tile2(const bf16_t* A, int strideA,
                                         const bf16_t* W, int WNt,
                                         int mt, int ntp, int Kt, int lane,
                                         v8f& acc0, v8f& acc1) {
  for (int kt = 0; kt < Kt; ++kt) {
    v16bf a  = load_a_frag(A, strideA, mt, kt, lane);
    v16bf b0 = load_b_frag(W, WNt, kt, 2 * ntp,     lane);
    v16bf b1 = load_b_frag(W, WNt, kt, 2 * ntp + 1, lane);
    acc0 = __builtin_amdgcn_wmma_f32_16x16x32_bf16(
        false, a, false, b0, (short)0, acc0, false, false);
    acc1 = __builtin_amdgcn_wmma_f32_16x16x32_bf16(
        false, a, false, b1, (short)0, acc1, false, false);
  }
}

// ---------------- fused per-batch GNN kernel ----------------
__global__ __launch_bounds__(256) void transition_gnn_kernel(
    const float* __restrict__ states, const int* __restrict__ action,
    const float* __restrict__ eb1, const float* __restrict__ eb2,
    const float* __restrict__ eg,  const float* __restrict__ ebt,
    const float* __restrict__ eb3,
    const float* __restrict__ nb1, const float* __restrict__ nb2,
    const float* __restrict__ ng,  const float* __restrict__ nbt,
    const float* __restrict__ nb3,
    const bf16_t* __restrict__ wEw1, const bf16_t* __restrict__ wEw2,
    const bf16_t* __restrict__ wEw3, const bf16_t* __restrict__ wNw1,
    const bf16_t* __restrict__ wNw2, const bf16_t* __restrict__ wNw3,
    float* __restrict__ out) {
  extern __shared__ char smem[];
  bf16_t* edge_in = (bf16_t*)(smem + LDS_EDGEIN);   // [96][64]
  bf16_t* act     = (bf16_t*)(smem + LDS_ACT);      // [96][512]
  float*  zbuf    = (float*)(smem + LDS_ZBUF);      // [96][512]
  float*  rmean   = (float*)(smem + LDS_RMEAN);     // [96]
  float*  rrstd   = (float*)(smem + LDS_RRSTD);     // [96]
  bf16_t* node_in = (bf16_t*)(smem + LDS_NODEIN);   // [16][576]
  bf16_t* nact    = (bf16_t*)(smem + LDS_NACT);     // [16][512]

  const int b    = blockIdx.x;
  const int tid  = threadIdx.x;
  const int wave = tid >> 5;
  const int lane = tid & 31;
  const int ggrp = lane >> 4;
  const int ln15 = lane & 15;
  const float* xs = states + (size_t)b * (NOBJ * DIN);

  // ---- Stage 0: edge inputs [96][64] bf16 (rows 90..95 zero) ----
  for (int idx = tid; idx < EPAD * 64; idx += 256) {
    int e = idx >> 6, c = idx & 63;
    float v = 0.0f;
    if (e < NEDGE) {
      int i = e / 9, jj = e - 9 * i;
      int j = jj + (jj >= i);
      v = (c < 32) ? xs[i * 32 + c] : xs[j * 32 + (c - 32)];
    }
    edge_in[idx] = (bf16_t)v;
  }
  __syncthreads();

  // ---- Stage 1: act = relu(edge_in @ ew1 + eb1)  [96x64]x[64x512] ----
  for (int job = wave; job < 6 * 16; job += WAVES) {
    int mt = job >> 4, ntp = job & 15;
    v8f a0 = {}, a1 = {};
    mm_tile2(edge_in, 64, wEw1, 32, mt, ntp, 2, lane, a0, a1);
    int n0 = ntp * 32 + ln15, n1 = n0 + 16;
    float bias0 = eb1[n0], bias1 = eb1[n1];
#pragma unroll
    for (int r = 0; r < 8; ++r) {
      int m = mt * 16 + r + 8 * ggrp;
      float v0 = a0[r] + bias0, v1 = a1[r] + bias1;
      act[m * HID + n0] = (bf16_t)(v0 > 0.0f ? v0 : 0.0f);
      act[m * HID + n1] = (bf16_t)(v1 > 0.0f ? v1 : 0.0f);
    }
  }
  __syncthreads();

  // ---- Stage 2: zbuf = act @ ew2 + eb2 ----
  for (int job = wave; job < 6 * 16; job += WAVES) {
    int mt = job >> 4, ntp = job & 15;
    v8f a0 = {}, a1 = {};
    mm_tile2(act, HID, wEw2, 32, mt, ntp, 16, lane, a0, a1);
    int n0 = ntp * 32 + ln15, n1 = n0 + 16;
    float bias0 = eb2[n0], bias1 = eb2[n1];
#pragma unroll
    for (int r = 0; r < 8; ++r) {
      int m = mt * 16 + r + 8 * ggrp;
      zbuf[m * HID + n0] = a0[r] + bias0;
      zbuf[m * HID + n1] = a1[r] + bias1;
    }
  }
  __syncthreads();
  // ---- LayerNorm stats (biased var, eps=1e-5) ----
  if (tid < EPAD) {
    const float* rowp = zbuf + tid * HID;
    float s = 0.0f, s2 = 0.0f;
    for (int nn = 0; nn < HID; ++nn) {
      int n = (nn + tid) & (HID - 1);          // stagger to avoid bank conflicts
      float v = rowp[n];
      s += v; s2 += v * v;
    }
    float m = s * (1.0f / HID);
    rmean[tid] = m;
    rrstd[tid] = rsqrtf(s2 * (1.0f / HID) - m * m + 1e-5f);
  }
  __syncthreads();
  // ---- normalize + affine + relu -> act bf16 ----
  for (int idx = tid; idx < EPAD * HID; idx += 256) {
    int m = idx >> 9, n = idx & (HID - 1);
    float v = (zbuf[idx] - rmean[m]) * rrstd[m] * eg[n] + ebt[n];
    act[idx] = (bf16_t)(v > 0.0f ? v : 0.0f);
  }
  __syncthreads();

  // ---- Stage 3: zbuf = act @ ew3 + eb3  (edge_attr) ----
  for (int job = wave; job < 6 * 16; job += WAVES) {
    int mt = job >> 4, ntp = job & 15;
    v8f a0 = {}, a1 = {};
    mm_tile2(act, HID, wEw3, 32, mt, ntp, 16, lane, a0, a1);
    int n0 = ntp * 32 + ln15, n1 = n0 + 16;
    float bias0 = eb3[n0], bias1 = eb3[n1];
#pragma unroll
    for (int r = 0; r < 8; ++r) {
      int m = mt * 16 + r + 8 * ggrp;
      zbuf[m * HID + n0] = a0[r] + bias0;
      zbuf[m * HID + n1] = a1[r] + bias1;
    }
  }
  __syncthreads();

  // ---- segment-sum over 9 contiguous edges/node + build node_in [16][576] ----
  int act_b = action[b];
  for (int idx = tid; idx < 16 * NIN_PAD; idx += 256) {
    int m = idx / NIN_PAD, c = idx - m * NIN_PAD;
    float v = 0.0f;
    if (m < NOBJ) {
      if (c < DIN) {
        v = xs[m * DIN + c];
      } else if (c < DIN + ADIM) {
        v = (act_b == (c - DIN)) ? 1.0f : 0.0f;
      } else if (c < DIN + ADIM + HID) {
        int n = c - (DIN + ADIM);
        float s = 0.0f;
#pragma unroll
        for (int r = 0; r < 9; ++r) s += zbuf[(m * 9 + r) * HID + n];
        v = s;
      }
    }
    node_in[idx] = (bf16_t)v;
  }
  __syncthreads();

  // ---- Stage 4: nact = relu(node_in @ nw1 + nb1)  [16x576]x[576x512] ----
  for (int job = wave; job < 16; job += WAVES) {
    int ntp = job;
    v8f a0 = {}, a1 = {};
    mm_tile2(node_in, NIN_PAD, wNw1, 32, 0, ntp, 18, lane, a0, a1);
    int n0 = ntp * 32 + ln15, n1 = n0 + 16;
    float bias0 = nb1[n0], bias1 = nb1[n1];
#pragma unroll
    for (int r = 0; r < 8; ++r) {
      int m = r + 8 * ggrp;
      float v0 = a0[r] + bias0, v1 = a1[r] + bias1;
      nact[m * HID + n0] = (bf16_t)(v0 > 0.0f ? v0 : 0.0f);
      nact[m * HID + n1] = (bf16_t)(v1 > 0.0f ? v1 : 0.0f);
    }
  }
  __syncthreads();

  // ---- Stage 5: zbuf[0..15] = nact @ nw2 + nb2 ----
  for (int job = wave; job < 16; job += WAVES) {
    int ntp = job;
    v8f a0 = {}, a1 = {};
    mm_tile2(nact, HID, wNw2, 32, 0, ntp, 16, lane, a0, a1);
    int n0 = ntp * 32 + ln15, n1 = n0 + 16;
    float bias0 = nb2[n0], bias1 = nb2[n1];
#pragma unroll
    for (int r = 0; r < 8; ++r) {
      int m = r + 8 * ggrp;
      zbuf[m * HID + n0] = a0[r] + bias0;
      zbuf[m * HID + n1] = a1[r] + bias1;
    }
  }
  __syncthreads();
  if (tid < 16) {
    const float* rowp = zbuf + tid * HID;
    float s = 0.0f, s2 = 0.0f;
    for (int nn = 0; nn < HID; ++nn) {
      int n = (nn + tid * 32) & (HID - 1);
      float v = rowp[n];
      s += v; s2 += v * v;
    }
    float m = s * (1.0f / HID);
    rmean[tid] = m;
    rrstd[tid] = rsqrtf(s2 * (1.0f / HID) - m * m + 1e-5f);
  }
  __syncthreads();
  for (int idx = tid; idx < 16 * HID; idx += 256) {
    int m = idx >> 9, n = idx & (HID - 1);
    float v = (zbuf[idx] - rmean[m]) * rrstd[m] * ng[n] + nbt[n];
    nact[idx] = (bf16_t)(v > 0.0f ? v : 0.0f);
  }
  __syncthreads();

  // ---- Stage 6: out = nact @ nw3 + nb3  [16x512]x[512x32], rows 0..9 ----
  for (int job = wave; job < 1; job += WAVES) {
    v8f a0 = {}, a1 = {};
    mm_tile2(nact, HID, wNw3, 2, 0, 0, 16, lane, a0, a1);
    int n0 = ln15, n1 = n0 + 16;
    float bias0 = nb3[n0], bias1 = nb3[n1];
#pragma unroll
    for (int r = 0; r < 8; ++r) {
      int m = r + 8 * ggrp;
      if (m < NOBJ) {
        out[(size_t)b * (NOBJ * DIN) + m * DIN + n0] = a0[r] + bias0;
        out[(size_t)b * (NOBJ * DIN) + m * DIN + n1] = a1[r] + bias1;
      }
    }
  }
}

// ---------------- host launch ----------------
extern "C" void kernel_launch(void* const* d_in, const int* in_sizes, int n_in,
                              void* d_out, int out_size, void* d_ws, size_t ws_size,
                              hipStream_t stream) {
  (void)in_sizes; (void)n_in; (void)out_size; (void)ws_size;
  const float* states = (const float*)d_in[0];
  const int*   action = (const int*)d_in[1];
  const float* ew1 = (const float*)d_in[2];
  const float* eb1 = (const float*)d_in[3];
  const float* ew2 = (const float*)d_in[4];
  const float* eb2 = (const float*)d_in[5];
  const float* eg  = (const float*)d_in[6];
  const float* ebt = (const float*)d_in[7];
  const float* ew3 = (const float*)d_in[8];
  const float* eb3 = (const float*)d_in[9];
  const float* nw1 = (const float*)d_in[10];
  const float* nb1 = (const float*)d_in[11];
  const float* nw2 = (const float*)d_in[12];
  const float* nb2 = (const float*)d_in[13];
  const float* ng  = (const float*)d_in[14];
  const float* nbt = (const float*)d_in[15];
  const float* nw3 = (const float*)d_in[16];
  const float* nb3 = (const float*)d_in[17];

  bf16_t* ws = (bf16_t*)d_ws;

  // Pre-swizzle weights into WMMA B-fragment order (bf16), recomputed each call.
  struct WDesc { const float* src; size_t off; int K, N, Nt, total; };
  const WDesc wd[6] = {
    { ew1, OFF_EW1,  64, 512, 32,  2 * 32 * 512 },
    { ew2, OFF_EW2, 512, 512, 32, 16 * 32 * 512 },
    { ew3, OFF_EW3, 512, 512, 32, 16 * 32 * 512 },
    { nw1, OFF_NW1, 548, 512, 32, 18 * 32 * 512 },
    { nw2, OFF_NW2, 512, 512, 32, 16 * 32 * 512 },
    { nw3, OFF_NW3, 512,  32,  2, 16 *  2 * 512 },
  };
  for (int i = 0; i < 6; ++i) {
    int grid = (wd[i].total + 255) / 256;
    swizzle_w_kernel<<<grid, 256, 0, stream>>>(wd[i].src, ws + wd[i].off,
                                               wd[i].K, wd[i].N, wd[i].Nt,
                                               wd[i].total);
  }

  // Allow >64KB dynamic LDS (CDNA5: 320KB per workgroup).
  (void)hipFuncSetAttribute((const void*)transition_gnn_kernel,
                            hipFuncAttributeMaxDynamicSharedMemorySize, LDS_BYTES);

  transition_gnn_kernel<<<NB, 256, LDS_BYTES, stream>>>(
      states, action, eb1, eb2, eg, ebt, eb3, nb1, nb2, ng, nbt, nb3,
      ws + OFF_EW1, ws + OFF_EW2, ws + OFF_EW3,
      ws + OFF_NW1, ws + OFF_NW2, ws + OFF_NW3,
      (float*)d_out);
}
